// LSTMModelModified_57011395887648
// MI455X (gfx1250) — matrix operand order repaired and checked by reading, not compiled, not run on hardware
//
#include <hip/hip_runtime.h>
#include <hip/hip_bf16.h>

typedef __attribute__((ext_vector_type(16))) _Float16 v16h;
typedef __attribute__((ext_vector_type(8)))  float    v8f;
typedef __attribute__((ext_vector_type(4)))  float    v4f;

#define B_SZ 1024
#define T_SZ 256
#define D_SZ 64
#define H_SZ 128
#define G_SZ 512           // 4*H
#define K0_SZ 192          // D + H   (layer 0 fused panel)
#define K1_SZ 256          // H + H   (layer 1 fused panel)
#define A0_PITCH 200
#define A1_PITCH 264
#define G_PITCH 516
#define LN_EPS 1e-5f

__device__ __forceinline__ float sigmoidf_fast(float x){ return 1.0f/(1.0f + __expf(-x)); }
__device__ __forceinline__ float tanhf_fast(float x){ return 1.0f - 2.0f/(__expf(2.0f*x) + 1.0f); }

#define WMMA_F16(A, Bm, C) \
  __builtin_amdgcn_wmma_f32_16x16x32_f16(false, (A), false, (Bm), (short)0, (C), false, false)

// ---------------------------------------------------------------------------
// Phase A: pack weights to f16 in concatenated [N=512][K] layout; merge biases.
// ---------------------------------------------------------------------------
__global__ __launch_bounds__(256) void prep_kernel(
    const float* __restrict__ Wih0, const float* __restrict__ Whh0,
    const float* __restrict__ bih0, const float* __restrict__ bhh0,
    const float* __restrict__ Wih1, const float* __restrict__ Whh1,
    const float* __restrict__ bih1, const float* __restrict__ bhh1,
    _Float16* __restrict__ W0cat, _Float16* __restrict__ W1cat,
    float* __restrict__ b0, float* __restrict__ b1)
{
  int i = blockIdx.x * blockDim.x + threadIdx.x;
  if (i < G_SZ * K0_SZ) {
    int n = i / K0_SZ, k = i % K0_SZ;
    float v = (k < D_SZ) ? Wih0[n*D_SZ + k] : Whh0[n*H_SZ + (k - D_SZ)];
    W0cat[i] = (_Float16)v;
  }
  if (i < G_SZ * K1_SZ) {
    int n = i / K1_SZ, k = i % K1_SZ;
    float v = (k < H_SZ) ? Wih1[n*H_SZ + k] : Whh1[n*H_SZ + (k - H_SZ)];
    W1cat[i] = (_Float16)v;
  }
  if (i < G_SZ) { b0[i] = bih0[i] + bhh0[i]; b1[i] = bih1[i] + bhh1[i]; }
}

// load one f16 B/A fragment (16 halves) from two 16B chunks
__device__ __forceinline__ v16h load_frag16(const _Float16* p0, const _Float16* p1) {
  v16h r; v4f* rv = (v4f*)&r;
  rv[0] = *(const v4f*)p0;
  rv[1] = *(const v4f*)p1;
  return r;
}

// ---------------------------------------------------------------------------
// Phase B: fused two-layer LSTM. One WG (512 thr = 16 waves) per 16 batch
// rows; h/c state in LDS across all 256 steps. Weight fragments for the
// recurrent parts are explicitly register-resident (192 VGPRs, under the
// 256-VGPR/wave cap at 16 waves/WGP); the small layer-0 x-part weights are
// re-read from L2 each step behind an anti-hoist opaque offset (keeps the
// global address space -> global_load_b128, tracked by LOADcnt only).
// ---------------------------------------------------------------------------
__global__ __launch_bounds__(512) void lstm_fused_kernel(
    const float* __restrict__ x,
    const _Float16* __restrict__ W0cat, const _Float16* __restrict__ W1cat,
    const float* __restrict__ b0, const float* __restrict__ b1,
    float* __restrict__ h2last)
{
  __shared__ __align__(16) _Float16 A0[16][A0_PITCH];   // [x_t | h0] panel
  __shared__ __align__(16) _Float16 A1[16][A1_PITCH];   // [h1_t | h1prev] panel
  __shared__ __align__(16) float    graw[16][G_PITCH];  // raw gates
  __shared__ __align__(16) float    c0s[16][H_SZ];
  __shared__ __align__(16) float    c1s[16][H_SZ];

  const int tid  = threadIdx.x;
  const int wave = tid >> 5;
  const int lane = tid & 31;
  const int m0   = blockIdx.x * 16;
  const int hi   = lane >> 4;
  const int ln   = lane & 15;
  const int nb0  = wave * 32;

  for (int idx = tid; idx < 16*H_SZ; idx += 512) {
    int m = idx >> 7, n = idx & 127;
    c0s[m][n] = 0.f; c1s[m][n] = 0.f;
    A0[m][D_SZ + n] = (_Float16)0.f;
    A1[m][H_SZ + n] = (_Float16)0.f;
  }
  __syncthreads();

  // per-wave biases
  const float bv00 = b0[nb0 + ln],      bv01 = b0[nb0 + 16 + ln];
  const float bv10 = b1[nb0 + ln],      bv11 = b1[nb0 + 16 + ln];

  // B-operand row bases: lane n holds weight row n (16 consecutive K halves)
  const _Float16* w0r0 = W0cat + (size_t)(nb0 + ln)      * K0_SZ + hi*16;
  const _Float16* w0r1 = W0cat + (size_t)(nb0 + 16 + ln) * K0_SZ + hi*16;
  const _Float16* w1r0 = W1cat + (size_t)(nb0 + ln)      * K1_SZ + hi*16;
  const _Float16* w1r1 = W1cat + (size_t)(nb0 + 16 + ln) * K1_SZ + hi*16;

  // register-cached weight fragments: layer-0 h-part (k=64..191) + all layer-1
  v16h w0h[2][4];
  v16h w1c[2][8];
  #pragma unroll
  for (int kk = 0; kk < 4; kk++) {
    w0h[0][kk] = load_frag16(w0r0 + (kk+2)*32, w0r0 + (kk+2)*32 + 8);
    w0h[1][kk] = load_frag16(w0r1 + (kk+2)*32, w0r1 + (kk+2)*32 + 8);
  }
  #pragma unroll
  for (int kk = 0; kk < 8; kk++) {
    w1c[0][kk] = load_frag16(w1r0 + kk*32, w1r0 + kk*32 + 8);
    w1c[1][kk] = load_frag16(w1r1 + kk*32, w1r1 + kk*32 + 8);
  }

  // A-operand row bases in LDS
  const _Float16* arow0 = &A0[0][0] + ln*A0_PITCH + hi*16;
  const _Float16* arow1 = &A1[0][0] + ln*A1_PITCH + hi*16;

  for (int t = 0; t < T_SZ; t++) {
    // prefetch next timestep's x rows into cache (global_prefetch_b8)
    if (t + 1 < T_SZ && tid < 32) {
      int r = tid >> 1, half = (tid & 1) << 5;
      __builtin_prefetch(&x[((size_t)(m0 + r)*T_SZ + (t + 1))*D_SZ + half], 0, 3);
    }
    // stage x_t (f32 -> f16) into panel columns 0..63
    for (int idx = tid; idx < 16*D_SZ; idx += 512) {
      int r = idx >> 6, k = idx & 63;
      A0[r][k] = (_Float16) x[((size_t)(m0 + r)*T_SZ + t)*D_SZ + k];
    }
    __syncthreads();

    // ---------------- layer-0 gate GEMM: [16 x 32] = A0[16xK0] * W0^T ------
    {
      v8f acc0 = {bv00,bv00,bv00,bv00,bv00,bv00,bv00,bv00};
      v8f acc1 = {bv01,bv01,bv01,bv01,bv01,bv01,bv01,bv01};

      // x-part weights: opaque zero offset defeats LICM (loads stay in-loop,
      // L2 hit) while preserving the global address space of the pointers.
      unsigned int ofs0 = 0;
      asm volatile("" : "+v"(ofs0));
      const _Float16* p0 = w0r0 + ofs0;
      const _Float16* p1 = w0r1 + ofs0;

      #pragma unroll
      for (int kk = 0; kk < 2; kk++) {
        v16h a   = load_frag16(arow0 + kk*32, arow0 + kk*32 + 8);
        v16h bf0 = load_frag16(p0 + kk*32, p0 + kk*32 + 8);
        v16h bf1 = load_frag16(p1 + kk*32, p1 + kk*32 + 8);
        acc0 = WMMA_F16(a, bf0, acc0);
        acc1 = WMMA_F16(a, bf1, acc1);
      }
      #pragma unroll
      for (int kk = 0; kk < 4; kk++) {
        v16h a = load_frag16(arow0 + (kk+2)*32, arow0 + (kk+2)*32 + 8);
        acc0 = WMMA_F16(a, w0h[0][kk], acc0);
        acc1 = WMMA_F16(a, w0h[1][kk], acc1);
      }
      #pragma unroll
      for (int r = 0; r < 8; r++) {
        graw[r + hi*8][nb0 + ln]      = acc0[r];
        graw[r + hi*8][nb0 + 16 + ln] = acc1[r];
      }
    }
    __syncthreads();

    // ---------------- layer-0 gate mix (i,f,g,o) ---------------------------
    for (int idx = tid; idx < 16*H_SZ; idx += 512) {
      int m = idx >> 7, n = idx & 127;
      float ig = sigmoidf_fast(graw[m][n]);
      float fg = sigmoidf_fast(graw[m][n + 128]);
      float gg = tanhf_fast   (graw[m][n + 256]);
      float og = sigmoidf_fast(graw[m][n + 384]);
      float c  = fg * c0s[m][n] + ig * gg;
      c0s[m][n] = c;
      float h  = og * tanhf_fast(c);
      A0[m][D_SZ + n] = (_Float16)h;   // recurrent input, next step layer 0
      A1[m][n]        = (_Float16)h;   // feed-forward input, layer 1 now
    }
    __syncthreads();

    // ---------------- layer-1 gate GEMM: [16 x 32] = A1[16x256] * W1^T -----
    {
      v8f acc0 = {bv10,bv10,bv10,bv10,bv10,bv10,bv10,bv10};
      v8f acc1 = {bv11,bv11,bv11,bv11,bv11,bv11,bv11,bv11};
      #pragma unroll
      for (int kk = 0; kk < 8; kk++) {
        v16h a = load_frag16(arow1 + kk*32, arow1 + kk*32 + 8);
        acc0 = WMMA_F16(a, w1c[0][kk], acc0);
        acc1 = WMMA_F16(a, w1c[1][kk], acc1);
      }
      #pragma unroll
      for (int r = 0; r < 8; r++) {
        graw[r + hi*8][nb0 + ln]      = acc0[r];
        graw[r + hi*8][nb0 + 16 + ln] = acc1[r];
      }
    }
    __syncthreads();

    // ---------------- layer-1 gate mix -------------------------------------
    for (int idx = tid; idx < 16*H_SZ; idx += 512) {
      int m = idx >> 7, n = idx & 127;
      float ig = sigmoidf_fast(graw[m][n]);
      float fg = sigmoidf_fast(graw[m][n + 128]);
      float gg = tanhf_fast   (graw[m][n + 256]);
      float og = sigmoidf_fast(graw[m][n + 384]);
      float c  = fg * c1s[m][n] + ig * gg;
      c1s[m][n] = c;
      float h  = og * tanhf_fast(c);
      A1[m][H_SZ + n] = (_Float16)h;
      if (t == T_SZ - 1) h2last[(size_t)(m0 + m)*H_SZ + n] = h;
    }
    __syncthreads();
  }
}

// ---------------------------------------------------------------------------
// Phase C: LayerNorm + FC(128->64) + ReLU + FC(64->1). One wave32 per row.
// ---------------------------------------------------------------------------
__global__ __launch_bounds__(256) void head_kernel(
    const float* __restrict__ h2last,
    const float* __restrict__ ln_g, const float* __restrict__ ln_b,
    const float* __restrict__ fc1_w, const float* __restrict__ fc1_b,
    const float* __restrict__ fc2_w, const float* __restrict__ fc2_b,
    float* __restrict__ out)
{
  __shared__ float ybuf[8][H_SZ];
  const int wave = threadIdx.x >> 5;
  const int lane = threadIdx.x & 31;
  const int b = blockIdx.x * 8 + wave;

  float v[4];
  #pragma unroll
  for (int j = 0; j < 4; j++) v[j] = h2last[(size_t)b*H_SZ + j*32 + lane];

  float s = v[0] + v[1] + v[2] + v[3];
  #pragma unroll
  for (int m = 16; m >= 1; m >>= 1) s += __shfl_xor(s, m);
  float mu = s * (1.0f / H_SZ);

  float q = 0.f;
  #pragma unroll
  for (int j = 0; j < 4; j++) { float d = v[j] - mu; q += d*d; }
  #pragma unroll
  for (int m = 16; m >= 1; m >>= 1) q += __shfl_xor(q, m);
  float rs = rsqrtf(q * (1.0f / H_SZ) + LN_EPS);

  #pragma unroll
  for (int j = 0; j < 4; j++) {
    int n = j*32 + lane;
    ybuf[wave][n] = (v[j] - mu) * rs * ln_g[n] + ln_b[n];
  }
  __syncthreads();

  float o2 = 0.f;
  #pragma unroll
  for (int kk = 0; kk < 2; kk++) {
    int k = kk*32 + lane;
    float a = fc1_b[k];
    for (int n = 0; n < H_SZ; n++) a += ybuf[wave][n] * fc1_w[k*H_SZ + n];
    o2 += fmaxf(a, 0.f) * fc2_w[k];
  }
  #pragma unroll
  for (int m = 16; m >= 1; m >>= 1) o2 += __shfl_xor(o2, m);
  if (lane == 0) out[b] = o2 + fc2_b[0];
}

// ---------------------------------------------------------------------------
extern "C" void kernel_launch(void* const* d_in, const int* in_sizes, int n_in,
                              void* d_out, int out_size, void* d_ws, size_t ws_size,
                              hipStream_t stream)
{
  (void)in_sizes; (void)n_in; (void)out_size; (void)ws_size;

  const float* x    = (const float*)d_in[0];
  const float* Wih0 = (const float*)d_in[1];
  const float* Whh0 = (const float*)d_in[2];
  const float* bih0 = (const float*)d_in[3];
  const float* bhh0 = (const float*)d_in[4];
  const float* Wih1 = (const float*)d_in[5];
  const float* Whh1 = (const float*)d_in[6];
  const float* bih1 = (const float*)d_in[7];
  const float* bhh1 = (const float*)d_in[8];
  const float* ln_g = (const float*)d_in[9];
  const float* ln_b = (const float*)d_in[10];
  const float* fc1_w = (const float*)d_in[11];
  const float* fc1_b = (const float*)d_in[12];
  const float* fc2_w = (const float*)d_in[13];
  const float* fc2_b = (const float*)d_in[14];
  float* out = (float*)d_out;

  // workspace layout (bytes): W0cat f16 192KB | W1cat f16 256KB | b0 2KB | b1 2KB | h2last 512KB
  char* ws = (char*)d_ws;
  _Float16* W0cat = (_Float16*)(ws);
  _Float16* W1cat = (_Float16*)(ws + 196608);
  float* b0       = (float*)(ws + 458752);
  float* b1       = (float*)(ws + 460800);
  float* h2last   = (float*)(ws + 462848);

  prep_kernel<<<512, 256, 0, stream>>>(Wih0, Whh0, bih0, bhh0,
                                       Wih1, Whh1, bih1, bhh1,
                                       W0cat, W1cat, b0, b1);
  lstm_fused_kernel<<<B_SZ/16, 512, 0, stream>>>(x, W0cat, W1cat, b0, b1, h2last);
  head_kernel<<<B_SZ/8, 256, 0, stream>>>(h2last, ln_g, ln_b,
                                          fc1_w, fc1_b, fc2_w, fc2_b, out);
}
